// GroupedQueryAttentionLayer_86234353369308
// MI455X (gfx1250) — matrix operand order repaired and checked by reading, not compiled, not run on hardware
//
#include <hip/hip_runtime.h>
#include <hip/hip_bf16.h>
#include <cstdint>

typedef __attribute__((ext_vector_type(16))) _Float16 v16h;
typedef __attribute__((ext_vector_type(8)))  _Float16 v8h;
typedef __attribute__((ext_vector_type(8)))  float    v8f;
typedef __attribute__((ext_vector_type(4)))  int      v4i;

static __device__ __forceinline__ v16h cat8(v8h lo, v8h hi) {
  return __builtin_shufflevector(lo, hi, 0,1,2,3,4,5,6,7,8,9,10,11,12,13,14,15);
}
static __device__ __forceinline__ v8h ld8(const _Float16* p) { return *(const v8h*)p; }

static __device__ __forceinline__ v8f wmma16(v16h a, v16h b, v8f c) {
  // D = A(16x32 f16) * B(32x16 f16) + C(16x16 f32)
  return __builtin_amdgcn_wmma_f32_16x16x32_f16(false, a, false, b, (short)0, c, false, false);
}

static __device__ __forceinline__ float rmax16(float x) {
  x = fmaxf(x, __shfl_xor(x, 1, 32));
  x = fmaxf(x, __shfl_xor(x, 2, 32));
  x = fmaxf(x, __shfl_xor(x, 4, 32));
  x = fmaxf(x, __shfl_xor(x, 8, 32));
  return x;
}
static __device__ __forceinline__ float rsum16(float x) {
  x += __shfl_xor(x, 1, 32);
  x += __shfl_xor(x, 2, 32);
  x += __shfl_xor(x, 4, 32);
  x += __shfl_xor(x, 8, 32);
  return x;
}

// ---- gfx1250 async global->LDS copy (ASYNCcnt), guarded with fallback ----
#if __has_builtin(__builtin_amdgcn_global_load_async_to_lds_b128) && \
    __has_builtin(__builtin_amdgcn_s_wait_asynccnt)
#define HAVE_ASYNC_LDS 1
#define AS1 __attribute__((address_space(1)))
#define AS3 __attribute__((address_space(3)))
static __device__ __forceinline__ void async_cp16(const _Float16* g, _Float16* l) {
  __builtin_amdgcn_global_load_async_to_lds_b128((AS1 v4i*)g, (AS3 v4i*)l, 0, 0);
}
#else
#define HAVE_ASYNC_LDS 0
#endif

// ---------------- conversion kernels ----------------

__global__ void __launch_bounds__(256) cvt8_f32_f16(const float* __restrict__ in,
                                                    _Float16* __restrict__ out, int n) {
  int i = (blockIdx.x * 256 + threadIdx.x) * 8;
  if (i >= n) return;
  float4 a = *(const float4*)(in + i);
  float4 b = *(const float4*)(in + i + 4);
  v8h o = { (_Float16)a.x, (_Float16)a.y, (_Float16)a.z, (_Float16)a.w,
            (_Float16)b.x, (_Float16)b.y, (_Float16)b.z, (_Float16)b.w };
  *(v8h*)(out + i) = o;
}

// Wt[n][k] = (f16) W[k][n]   (W is [K][N] row-major)
__global__ void __launch_bounds__(256) cvtT_f32_f16(const float* __restrict__ W,
                                                    _Float16* __restrict__ Wt, int K, int N) {
  int id = blockIdx.x * 256 + threadIdx.x;
  int n = id % N, k = id / N;
  if (k >= K) return;
  Wt[(size_t)n * K + k] = (_Float16)W[(size_t)k * N + n];
}

// ---------------- WMMA GEMM  C[M,N] = A[M,K] * Bt[N,K]^T ----------------
// OUT_MODE 0: f16 row-major; 1: f16 written into Vt[b][g][e][t] layout;
// 2: f32 row-major + residual add.
template <int OUT_MODE>
__global__ void __launch_bounds__(128) gemm_wmma(
    const _Float16* __restrict__ A, const _Float16* __restrict__ Bt,
    void* __restrict__ Cv, const float* __restrict__ resid,
    int M, int N, int K, float scale)
{
  __shared__ _Float16 sA[2][64][40];   // double-buffered 64 rows x 32 k (pad 40)
  __shared__ _Float16 sB[2][64][40];

  const int tid = threadIdx.x;
  const int wave = tid >> 5, lane = tid & 31, lm = lane & 15, hi = lane >> 4;
  const int m0 = blockIdx.y * 64, n0 = blockIdx.x * 64;

  v8f acc[4] = {};

  const int row = tid >> 1;          // 0..63
  const int seg = (tid & 1) << 4;    // 0 or 16 halves
  const _Float16* ga = A  + (size_t)(m0 + row) * K + seg;
  const _Float16* gb = Bt + (size_t)(n0 + row) * K + seg;

  auto compute = [&](int buf) {
    // A fragment: lane(m=lm,hi): halves = A[m][8hi+0..7], A[m][16+8hi+0..7]
    v16h af = cat8(ld8(&sA[buf][wave * 16 + lm][hi * 8]),
                   ld8(&sA[buf][wave * 16 + lm][hi * 8 + 16]));
#pragma unroll
    for (int j = 0; j < 4; ++j) {
      // B fragment: lane(n=lm,hi): halves = Bt[n][8hi+0..7], Bt[n][16+8hi+0..7]
      v16h bf = cat8(ld8(&sB[buf][j * 16 + lm][hi * 8]),
                     ld8(&sB[buf][j * 16 + lm][hi * 8 + 16]));
      acc[j] = wmma16(af, bf, acc[j]);
    }
  };

#if HAVE_ASYNC_LDS
  auto stage = [&](int buf, int kk) {  // 4 async b128 copies / thread
    async_cp16(ga + kk,     &sA[buf][row][seg]);
    async_cp16(ga + kk + 8, &sA[buf][row][seg + 8]);
    async_cp16(gb + kk,     &sB[buf][row][seg]);
    async_cp16(gb + kk + 8, &sB[buf][row][seg + 8]);
  };
  stage(0, 0);
  int buf = 0;
  for (int kk = 0; kk < K; kk += 32) {
    if (kk + 32 < K) {
      stage(buf ^ 1, kk + 32);                 // issue next tile async
      __builtin_amdgcn_s_wait_asynccnt(4);     // previous tile done (in-order)
    } else {
      __builtin_amdgcn_s_wait_asynccnt(0);
    }
    __syncthreads();
    compute(buf);
    __syncthreads();                           // reads done before buf is re-staged
    buf ^= 1;
  }
#else
  for (int kk = 0; kk < K; kk += 32) {
    v8h a0 = ld8(ga + kk), a1 = ld8(ga + kk + 8);
    v8h b0 = ld8(gb + kk), b1 = ld8(gb + kk + 8);
    *(v8h*)&sA[0][row][seg]     = a0;  *(v8h*)&sA[0][row][seg + 8] = a1;
    *(v8h*)&sB[0][row][seg]     = b0;  *(v8h*)&sB[0][row][seg + 8] = b1;
    __syncthreads();
    compute(0);
    __syncthreads();
  }
#endif

#pragma unroll
  for (int j = 0; j < 4; ++j) {
#pragma unroll
    for (int v = 0; v < 8; ++v) {
      int r = m0 + wave * 16 + v + 8 * hi;     // C layout: VGPR v -> row v + 8*hi
      int c = n0 + j * 16 + lm;                // lane -> column
      float val = acc[j][v] * scale;
      if (OUT_MODE == 0) {
        ((_Float16*)Cv)[(size_t)r * N + c] = (_Float16)val;
      } else if (OUT_MODE == 1) {
        int bb = r >> 11, t = r & 2047;        // M = b*2048 + t
        int gg = c >> 6,  e = c & 63;          // N = g*64 + e
        ((_Float16*)Cv)[((size_t)(bb * 4 + gg) * 64 + e) * 2048 + t] = (_Float16)val;
      } else {
        size_t idx = (size_t)r * N + c;
        ((float*)Cv)[idx] = val + resid[idx];
      }
    }
  }
}

// ---------------- flash attention (grouped query) ----------------
// Q:  [b,s,h,e] f16 (pre-scaled by 1/8), Kb: [b,t,g,e] f16, Vt: [b,g,e,t] f16
// Ab: [b,s,h,e] f16 output.
__global__ void __launch_bounds__(128) attn_wmma(
    const _Float16* __restrict__ Q, const _Float16* __restrict__ Kb,
    const _Float16* __restrict__ Vt, _Float16* __restrict__ Ab)
{
  __shared__ _Float16 sP[4][16][40];  // per-wave P staging (16 rows x 32 t, pad 40)

  const int tid = threadIdx.x, wave = tid >> 5, lane = tid & 31;
  const int lm = lane & 15, hi = lane >> 4;
  const int b = blockIdx.z, h = blockIdx.y, g = h >> 2;
  const int s0 = blockIdx.x * 64 + wave * 16;

  const _Float16* qrow = Q  + (size_t)(b * 2048 + s0 + lm) * 1024 + h * 64;
  const _Float16* kb   = Kb + (size_t)(b * 2048) * 256 + g * 64;
  const _Float16* vtb  = Vt + (size_t)((b * 4 + g) * 64) * 2048;

  // Q as A-operand, two 16x32 e-chunks, resident for the whole t loop
  v16h qf[2];
#pragma unroll
  for (int c = 0; c < 2; ++c) {
    const _Float16* p = qrow + c * 32 + hi * 8;
    qf[c] = cat8(ld8(p), ld8(p + 16));
  }

  v8f o[4] = {};
  float mrow[8], lrow[8];
#pragma unroll
  for (int v = 0; v < 8; ++v) { mrow[v] = -1e30f; lrow[v] = 0.f; }
  const float LOG2E = 1.4426950408889634f;

  for (int tt = 0; tt < 2048; tt += 32) {
    if (tt + 32 < 2048) {  // keep next K/V tiles warm (global_prefetch_b8)
      __builtin_prefetch(kb + (size_t)(tt + 32 + lm) * 256, 0, 3);
      __builtin_prefetch(vtb + (size_t)lm * 2048 + tt + 32, 0, 3);
    }
    // ---- S = Q * K^T for 32 keys (two 16-col fragments) ----
    v8f sf0 = {}, sf1 = {};
#pragma unroll
    for (int c = 0; c < 2; ++c) {
      // B-operand for S: halves = K[tt + n][e0 + 8hi + 0..7], K[tt + n][e0 + 16 + 8hi + 0..7]
      const _Float16* k0 = kb + (size_t)(tt + lm) * 256 + c * 32 + hi * 8;
      const _Float16* k1 = k0 + 16 * 256;
      v16h bf0 = cat8(ld8(k0), ld8(k0 + 16));
      v16h bf1 = cat8(ld8(k1), ld8(k1 + 16));
      sf0 = wmma16(qf[c], bf0, sf0);
      sf1 = wmma16(qf[c], bf1, sf1);
    }
    // ---- online softmax over the 32 new columns ----
#pragma unroll
    for (int v = 0; v < 8; ++v) {
      float m = rmax16(fmaxf(sf0[v], sf1[v]));
      float mnew = fmaxf(mrow[v], m);
      float alpha = exp2f((mrow[v] - mnew) * LOG2E);
      mrow[v] = mnew;
      float p0 = exp2f((sf0[v] - mnew) * LOG2E);
      float p1 = exp2f((sf1[v] - mnew) * LOG2E);
      lrow[v] = lrow[v] * alpha + rsum16(p0 + p1);
#pragma unroll
      for (int j = 0; j < 4; ++j) o[j][v] *= alpha;
      int r = v + 8 * hi;                       // this lane owns rows v + 8*hi
      sP[wave][r][lm]      = (_Float16)p0;      // column = lane
      sP[wave][r][16 + lm] = (_Float16)p1;
    }
    __syncthreads();
    // ---- O += P * V  (P reloaded from LDS in A-layout) ----
    v16h pf = cat8(ld8(&sP[wave][lm][hi * 8]), ld8(&sP[wave][lm][hi * 8 + 16]));
#pragma unroll
    for (int j = 0; j < 4; ++j) {
      // B-operand for P*V: halves = Vt[e][tt+8hi+0..7], Vt[e][tt+16+8hi+0..7]
      const _Float16* vp = vtb + (size_t)(j * 16 + lm) * 2048 + tt + hi * 8;
      v16h vf = cat8(ld8(vp), ld8(vp + 16));
      o[j] = wmma16(pf, vf, o[j]);
    }
    __syncthreads();
  }

  // normalize and store O
#pragma unroll
  for (int v = 0; v < 8; ++v) {
    float inv = 1.0f / lrow[v];
    int srow = s0 + v + 8 * hi;
    _Float16* outp = Ab + (size_t)(b * 2048 + srow) * 1024 + h * 64 + lm;
#pragma unroll
    for (int j = 0; j < 4; ++j) outp[j * 16] = (_Float16)(o[j][v] * inv);
  }
}

// ---------------- launcher ----------------

extern "C" void kernel_launch(void* const* d_in, const int* in_sizes, int n_in,
                              void* d_out, int out_size, void* d_ws, size_t ws_size,
                              hipStream_t stream) {
  const float* X  = (const float*)d_in[0];   // [2,2048,1024]
  const float* Wq = (const float*)d_in[1];   // [1024,1024]
  const float* Wk = (const float*)d_in[2];   // [1024,256]
  const float* Wv = (const float*)d_in[3];   // [1024,256]
  const float* Wp = (const float*)d_in[4];   // [1024,1024]

  char* ws = (char*)d_ws;
  const size_t MB = 1024 * 1024;
  _Float16* Xh  = (_Float16*)(ws + 0);                   // 8 MB  [4096,1024]
  _Float16* Wqt = (_Float16*)(ws + 8 * MB);              // 2 MB  [1024,1024] (N,K)
  _Float16* Wkt = (_Float16*)(ws + 10 * MB);             // .5 MB [256,1024]
  _Float16* Wvt = (_Float16*)(ws + 10 * MB + 512 * 1024);// .5 MB [256,1024]
  _Float16* Wpt = (_Float16*)(ws + 11 * MB);             // 2 MB  [1024,1024]
  _Float16* Qb  = (_Float16*)(ws + 13 * MB);             // 8 MB  [b,s,h,e] (scaled)
  _Float16* Kbf = (_Float16*)(ws + 21 * MB);             // 2 MB  [b,t,g,e]
  _Float16* Vtf = (_Float16*)(ws + 23 * MB);             // 2 MB  [b,g,e,t]
  _Float16* Ab  = (_Float16*)(ws + 25 * MB);             // 8 MB  [b,s,h,e]

  cvt8_f32_f16<<<2048, 256, 0, stream>>>(X, Xh, 4096 * 1024);
  cvtT_f32_f16<<<4096, 256, 0, stream>>>(Wq, Wqt, 1024, 1024);
  cvtT_f32_f16<<<1024, 256, 0, stream>>>(Wk, Wkt, 1024, 256);
  cvtT_f32_f16<<<1024, 256, 0, stream>>>(Wv, Wvt, 1024, 256);
  cvtT_f32_f16<<<4096, 256, 0, stream>>>(Wp, Wpt, 1024, 1024);

  // Q (scale = 1/sqrt(64) folded into epilogue), K, V (V stored transposed)
  gemm_wmma<0><<<dim3(16, 64), 128, 0, stream>>>(Xh, Wqt, Qb,  nullptr, 4096, 1024, 1024, 0.125f);
  gemm_wmma<0><<<dim3(4,  64), 128, 0, stream>>>(Xh, Wkt, Kbf, nullptr, 4096, 256,  1024, 1.0f);
  gemm_wmma<1><<<dim3(4,  64), 128, 0, stream>>>(Xh, Wvt, Vtf, nullptr, 4096, 256,  1024, 1.0f);

  attn_wmma<<<dim3(32, 16, 2), 128, 0, stream>>>(Qb, Kbf, Vtf, Ab);

  // P = A @ Wpost + X  -> fp32 output
  gemm_wmma<2><<<dim3(16, 64), 128, 0, stream>>>(Ab, Wpt, d_out, X, 4096, 1024, 1024, 1.0f);
}
